// MTRNN_10505490006709
// MI455X (gfx1250) — matrix operand order repaired and checked by reading, not compiled, not run on hardware
//
#include <hip/hip_runtime.h>
#include <hip/hip_fp16.h>

// ---------------- problem dims ----------------
#define I_DIM 512
#define H_DIM 1024
#define B_DIM 64
#define T_DIM 256
#define G_DIM 4096          // 4*H gate width

// ---------------- kernel config ----------------
#define NWG   32            // persistent workgroups; 32 WG * 8 waves * 16 cols = 4096 = G_DIM
#define NTHR  256           // 8 wave32 per WG
#define BK    128           // K-chunk staged in LDS per stage
#define LDSTRIDE (BK + 8)   // +8 f16 (=16B) pad -> 272B row stride, conflict-free ds_load_b128
#define BUFELEMS (B_DIM * LDSTRIDE)   // one LDS staging buffer (f16 elems)

typedef _Float16 half_t;
typedef __attribute__((ext_vector_type(16))) _Float16     v16h;
typedef __attribute__((ext_vector_type(8)))  float        v8f;
typedef __attribute__((ext_vector_type(4)))  unsigned int u32x4;

// scheduler shaping (no-op if builtin unavailable)
#if __has_builtin(__builtin_amdgcn_sched_group_barrier)
#define SGB(mask, size, id) __builtin_amdgcn_sched_group_barrier((mask), (size), (id))
#else
#define SGB(mask, size, id)
#endif

// -------- force true global_* ops (avoid flat_load's LOADcnt+DScnt coupling) --------
#define GAS __attribute__((address_space(1)))
__device__ __forceinline__ u32x4 gload_u32x4(const void* p) {
  return *(const GAS u32x4*)(unsigned long long)p;
}
__device__ __forceinline__ float gload_f32(const void* p) {
  return *(const GAS float*)(unsigned long long)p;
}
__device__ __forceinline__ void gstore_f32(void* p, float v) {
  *(GAS float*)(unsigned long long)p = v;
}
__device__ __forceinline__ void gstore_f16(void* p, half_t v) {
  *(GAS half_t*)(unsigned long long)p = v;
}

// -------- fast activations: v_exp_f32 + v_rcp_f32, no libm branches --------
__device__ __forceinline__ float fast_sigmoid(float x) {
  float e = __expf(-x);                       // x very negative -> e=inf -> rcp -> 0  (graceful)
  return __builtin_amdgcn_rcpf(1.0f + e);
}
__device__ __forceinline__ float fast_tanh(float x) {
  x = fminf(fmaxf(x, -15.0f), 15.0f);         // tanh saturated well before +-15; keeps exp finite
  float e = __expf(2.0f * x);
  return (e - 1.0f) * __builtin_amdgcn_rcpf(e + 1.0f);
}

struct Params {
  const half_t* x;                      // [B][T][I] f16
  const half_t* W0;                     // [4096][1536] f16  packed [Wih1|Whh1]
  const half_t* W1;                     // [4096][2048] f16
  const half_t* W2;
  const half_t* W3;
  const float*  bias;                   // [4][4096] f32 (bih+bhh)
  half_t*       hbf;                    // [4][B][H] f16 hidden (GEMM operand)
  float*        hf;                     // [4][B][H] f32 hidden (precision copy)
  float*        c;                      // [4][B][H] f32 cell state
  float*        G;                      // [B][4096] f32 gate scratch
  unsigned*     sync;                   // [cnt, gen] grid barrier
  float*        out;                    // [B][H] f32 = h4
};

// ---------------- grid-wide barrier (persistent kernel) ----------------
__device__ __forceinline__ void gsync(unsigned* sync) {
  __threadfence();          // make this WG's global stores visible device-wide
  __syncthreads();
  if (threadIdx.x == 0) {
    unsigned* cnt = sync;
    unsigned* gen = sync + 1;
    unsigned g = __hip_atomic_load(gen, __ATOMIC_ACQUIRE, __HIP_MEMORY_SCOPE_AGENT);
    unsigned arrived = __hip_atomic_fetch_add(cnt, 1u, __ATOMIC_ACQ_REL, __HIP_MEMORY_SCOPE_AGENT);
    if (arrived == NWG - 1) {
      __hip_atomic_store(cnt, 0u, __ATOMIC_RELAXED, __HIP_MEMORY_SCOPE_AGENT);
      __hip_atomic_fetch_add(gen, 1u, __ATOMIC_RELEASE, __HIP_MEMORY_SCOPE_AGENT);
    } else {
      while (__hip_atomic_load(gen, __ATOMIC_ACQUIRE, __HIP_MEMORY_SCOPE_AGENT) == g) {
        __builtin_amdgcn_s_sleep(2);
      }
    }
  }
  __syncthreads();
}

// ---------------- async DMA of one act chunk [64][BK] f16 into LDS ----------------
// lenA % BK == 0 and k0 % BK == 0  ->  each chunk reads from exactly one source,
// so a single uniform SADDR base + per-lane 32-bit offset (GVS mode) suffices.
__device__ __forceinline__ void stage_async(const half_t* __restrict__ srcA, int lenA, int strideA,
                                            const half_t* __restrict__ srcB,
                                            int k0, unsigned lds_byte_base)
{
  const int tid = threadIdx.x;
  const half_t* base;
  int stride, kloc;
  if (k0 < lenA) { base = srcA; stride = strideA; kloc = k0; }
  else           { base = srcB; stride = H_DIM;   kloc = k0 - lenA; }
  unsigned long long sbase = (unsigned long long)base;
#pragma unroll
  for (int j = 0; j < (B_DIM * BK) / (NTHR * 8); ++j) {   // 4 x b128 per thread
    int idx = j * NTHR + tid;            // 16B-chunk index; 16 chunks per row
    int m   = idx >> 4;
    int kk  = (idx & 15) << 3;
    unsigned loff = lds_byte_base + (unsigned)(m * LDSTRIDE + kk) * 2u;
    unsigned goff = (unsigned)(m * stride + kloc + kk) * 2u;
    asm volatile("global_load_async_to_lds_b128 %0, %1, %2"
                 :: "v"(loff), "v"(goff), "s"(sbase)
                 : "memory");
  }
}

// ---------------- GEMM: gates[64][4096] = act[64][K] @ W^T, f16 WMMA ----------------
// act row m = [srcA[m*strideA + 0..lenA) | srcB[m*H + 0..H)]
// W row-major [4096][K]: column n of the (K x N) B-operand is a contiguous row of W.
__device__ __forceinline__ void cell_gemm(const half_t* __restrict__ srcA, int lenA, int strideA,
                                          const half_t* __restrict__ srcB,
                                          const half_t* __restrict__ W, int K,
                                          float* __restrict__ G,
                                          half_t* lds)
{
  const int tid  = threadIdx.x;
  const int wave = tid >> 5;            // wave32
  const int lane = tid & 31;
  const int col  = lane & 15;
  const int sel  = lane >> 4;
  const int n0   = blockIdx.x * 128 + wave * 16;   // this wave's 16 gate columns

  v8f acc[4] = {};                      // 4 M-tiles (rows 0..63) x 16 cols, f32

  // B fragment base for this lane: row (n0+col) of W; lanes 0-15 K=0..7,16..23, 16-31 K=8..15,24..31
  const half_t* wrow = W + (size_t)(n0 + col) * (size_t)K + sel * 8;

  const unsigned lds0 = (unsigned)(unsigned long long)(void*)lds;   // LDS byte offset of buffer 0
  const int S = K / BK;

  stage_async(srcA, lenA, strideA, srcB, 0, lds0);      // prime pipeline -> buf0

  union Frag { v16h v; u32x4 q[2]; };

#pragma unroll 1
  for (int s = 0; s < S; ++s) {
    asm volatile("s_wait_asynccnt 0x0" ::: "memory");   // my chunk-s DMA landed
    __syncthreads();                                    // everyone's chunk s visible
    if (s + 1 < S)                                      // overlap next DMA with compute
      stage_async(srcA, lenA, strideA, srcB, (s + 1) * BK,
                  lds0 + (unsigned)(((s + 1) & 1) * BUFELEMS * 2));

    const half_t* buf = lds + (s & 1) * BUFELEMS;
    const int k0 = s * BK;

    if (s + 1 < S) __builtin_prefetch(wrow + k0 + BK, 0, 3);   // next weight chunk -> cache

    Frag bf[4];                                         // 8 global_load_b128
#pragma unroll
    for (int i = 0; i < 4; ++i) {
      const half_t* wp = wrow + k0 + i * 32;
      bf[i].q[0] = gload_u32x4(wp);
      bf[i].q[1] = gload_u32x4(wp + 16);
    }
    Frag af[4][4];                                      // 32 ds_load_b128
#pragma unroll
    for (int i = 0; i < 4; ++i) {
#pragma unroll
      for (int mt = 0; mt < 4; ++mt) {
        const half_t* ap = buf + (mt * 16 + col) * LDSTRIDE + i * 32 + sel * 8;
        af[i][mt].q[0] = *(const u32x4*)(ap);
        af[i][mt].q[1] = *(const u32x4*)(ap + 16);
      }
    }
#pragma unroll
    for (int i = 0; i < 4; ++i) {
#pragma unroll
      for (int mt = 0; mt < 4; ++mt) {
        acc[mt] = __builtin_amdgcn_wmma_f32_16x16x32_f16(
            false, af[i][mt].v, false, bf[i].v, (short)0, acc[mt], false, false);
      }
    }

    // ---- pin the stage schedule: depth-4 LDS->WMMA software pipeline ----
    // 8 VMEM reads (B frags) ; 8 DS reads (first 4 WMMAs' A frags) ;
    // then 12 x { 1 WMMA, 2 DS reads (operands 4 WMMAs ahead) } ; 4 WMMA drain.
    SGB(0x020, 8, 0);     // VMEM read x8
    SGB(0x100, 8, 0);     // DS read   x8
#pragma unroll
    for (int g = 0; g < 12; ++g) {
      SGB(0x008, 1, 0);   // WMMA      x1
      SGB(0x100, 2, 0);   // DS read   x2
    }
    SGB(0x008, 4, 0);     // WMMA      x4
  }

  // ---- store gates: C/D layout: VGPR r, lanes 0-15 -> M=r, lanes 16-31 -> M=r+8 ----
#pragma unroll
  for (int mt = 0; mt < 4; ++mt) {
#pragma unroll
    for (int r = 0; r < 8; ++r) {
      int m = mt * 16 + r + sel * 8;
      gstore_f32(&G[(size_t)m * G_DIM + n0 + col], acc[mt][r]);
    }
  }
}

// ---------------- elementwise LSTM cell update ----------------
__device__ __forceinline__ void cell_ew(const float* __restrict__ G,
                                        const float* __restrict__ bias,
                                        float* __restrict__ c,
                                        float* __restrict__ hf,
                                        half_t* __restrict__ hbf)
{
  int gtid = blockIdx.x * NTHR + threadIdx.x;
#pragma unroll 1
  for (int idx = gtid; idx < B_DIM * H_DIM; idx += NWG * NTHR) {
    int m = idx >> 10;
    int n = idx & (H_DIM - 1);
    const float* gm = G + (size_t)m * G_DIM + n;
    float gi = gload_f32(gm)             + gload_f32(bias + n);
    float gf = gload_f32(gm + H_DIM)     + gload_f32(bias + H_DIM + n);
    float gg = gload_f32(gm + 2 * H_DIM) + gload_f32(bias + 2 * H_DIM + n);
    float go = gload_f32(gm + 3 * H_DIM) + gload_f32(bias + 3 * H_DIM + n);
    float i = fast_sigmoid(gi);
    float f = fast_sigmoid(gf);
    float g = fast_tanh(gg);
    float o = fast_sigmoid(go);
    float cn = f * gload_f32(c + idx) + i * g;
    float hn = o * fast_tanh(cn);
    gstore_f32(c + idx, cn);
    gstore_f32(hf + idx, hn);
    gstore_f16(hbf + idx, (half_t)hn);
  }
}

// ---------------- persistent kernel: entire T=256 scan ----------------
__global__ __launch_bounds__(NTHR) void mtrnn_persistent(Params p)
{
  __shared__ half_t lds[2 * BUFELEMS];   // double-buffered act staging, 34816 B
  const half_t* Ws[4] = {p.W0, p.W1, p.W2, p.W3};

#pragma unroll 1
  for (int t = 0; t < T_DIM; ++t) {
#pragma unroll 1
    for (int cell = 0; cell < 4; ++cell) {
      if (cell == 1 && (t & 1)) continue;    // cell2: t%2==0
      if (cell == 2 && (t & 3)) continue;    // cell3: t%4==0
      if (cell == 3 && (t & 7)) continue;    // cell4: t%8==0

      const half_t* srcA;
      int lenA, strideA, K;
      if (cell == 0) {
        srcA = p.x + (size_t)t * I_DIM;      // x[m][t][:], row stride T*I
        lenA = I_DIM; strideA = T_DIM * I_DIM; K = I_DIM + H_DIM;
      } else {
        srcA = p.hbf + (size_t)(cell - 1) * B_DIM * H_DIM;
        lenA = H_DIM; strideA = H_DIM; K = 2 * H_DIM;
      }
      const half_t* srcB = p.hbf + (size_t)cell * B_DIM * H_DIM;   // recurrent h

      cell_gemm(srcA, lenA, strideA, srcB, Ws[cell], K, p.G, lds);
      gsync(p.sync);
      cell_ew(p.G, p.bias + cell * G_DIM,
              p.c  + (size_t)cell * B_DIM * H_DIM,
              p.hf + (size_t)cell * B_DIM * H_DIM,
              p.hbf + (size_t)cell * B_DIM * H_DIM);
      gsync(p.sync);
    }
  }

  // write h4 (f32) to output
  int gtid = blockIdx.x * NTHR + threadIdx.x;
  const float* h4 = p.hf + (size_t)3 * B_DIM * H_DIM;
#pragma unroll 1
  for (int idx = gtid; idx < B_DIM * H_DIM; idx += NWG * NTHR)
    gstore_f32(p.out + idx, gload_f32(h4 + idx));
}

// ---------------- prep kernels ----------------
__global__ void k_init(unsigned* sync, float* c, float* hf, half_t* hbf)
{
  size_t gtid   = (size_t)blockIdx.x * blockDim.x + threadIdx.x;
  size_t stride = (size_t)gridDim.x * blockDim.x;
  const size_t n = (size_t)4 * B_DIM * H_DIM;
  for (size_t i = gtid; i < n; i += stride) {
    c[i] = 0.0f; hf[i] = 0.0f; hbf[i] = (half_t)0.0f;
  }
  if (gtid < 2) sync[gtid] = 0u;
}

__global__ void k_cvt(const float* __restrict__ src, half_t* __restrict__ dst, size_t n)
{
  size_t gtid   = (size_t)blockIdx.x * blockDim.x + threadIdx.x;
  size_t stride = (size_t)gridDim.x * blockDim.x;
  for (size_t i = gtid; i < n; i += stride) dst[i] = (half_t)src[i];
}

// pack [Wih | Whh] rows into one f16 matrix [4096][K]; bias = bih + bhh
__global__ void k_pack(const float* __restrict__ Wih, const float* __restrict__ Whh,
                       const float* __restrict__ bih, const float* __restrict__ bhh,
                       half_t* __restrict__ Wd, float* __restrict__ bd,
                       int lenA, int K)
{
  size_t gtid   = (size_t)blockIdx.x * blockDim.x + threadIdx.x;
  size_t stride = (size_t)gridDim.x * blockDim.x;
  size_t n = (size_t)G_DIM * (size_t)K;
  for (size_t i = gtid; i < n; i += stride) {
    size_t row = i / (size_t)K;
    int    k   = (int)(i - row * (size_t)K);
    float v = (k < lenA) ? Wih[row * (size_t)lenA + k]
                         : Whh[row * (size_t)H_DIM + (k - lenA)];
    Wd[i] = (half_t)v;
  }
  for (size_t i = gtid; i < G_DIM; i += stride) bd[i] = bih[i] + bhh[i];
}

// ---------------- host launch ----------------
extern "C" void kernel_launch(void* const* d_in, const int* in_sizes, int n_in,
                              void* d_out, int out_size, void* d_ws, size_t ws_size,
                              hipStream_t stream)
{
  (void)in_sizes; (void)n_in; (void)out_size; (void)ws_size;

  const float* x = (const float*)d_in[0];
  const float* Wih[4] = {(const float*)d_in[1], (const float*)d_in[5],
                         (const float*)d_in[9], (const float*)d_in[13]};
  const float* Whh[4] = {(const float*)d_in[2], (const float*)d_in[6],
                         (const float*)d_in[10], (const float*)d_in[14]};
  const float* bih[4] = {(const float*)d_in[3], (const float*)d_in[7],
                         (const float*)d_in[11], (const float*)d_in[15]};
  const float* bhh[4] = {(const float*)d_in[4], (const float*)d_in[8],
                         (const float*)d_in[12], (const float*)d_in[16]};

  char* ws = (char*)d_ws;
  size_t off = 0;
  auto take = [&](size_t bytes) -> void* {
    void* p = ws + off;
    off += (bytes + 255) & ~(size_t)255;
    return p;
  };

  unsigned* sync = (unsigned*)take(256);
  half_t* xh = (half_t*)take((size_t)B_DIM * T_DIM * I_DIM * sizeof(half_t));
  half_t* W[4];
  const int Kd[4]   = {I_DIM + H_DIM, 2 * H_DIM, 2 * H_DIM, 2 * H_DIM};
  const int lenA[4] = {I_DIM, H_DIM, H_DIM, H_DIM};
  for (int cell = 0; cell < 4; ++cell)
    W[cell] = (half_t*)take((size_t)G_DIM * Kd[cell] * sizeof(half_t));
  float*  bias = (float*)take((size_t)4 * G_DIM * sizeof(float));
  float*  c    = (float*)take((size_t)4 * B_DIM * H_DIM * sizeof(float));
  float*  hf   = (float*)take((size_t)4 * B_DIM * H_DIM * sizeof(float));
  half_t* hbf  = (half_t*)take((size_t)4 * B_DIM * H_DIM * sizeof(half_t));
  float*  G    = (float*)take((size_t)B_DIM * G_DIM * sizeof(float));

  k_init<<<64, NTHR, 0, stream>>>(sync, c, hf, hbf);
  k_cvt<<<256, NTHR, 0, stream>>>(x, xh, (size_t)B_DIM * T_DIM * I_DIM);
  for (int cell = 0; cell < 4; ++cell)
    k_pack<<<256, NTHR, 0, stream>>>(Wih[cell], Whh[cell], bih[cell], bhh[cell],
                                     W[cell], bias + cell * G_DIM,
                                     lenA[cell], Kd[cell]);

  Params p;
  p.x = xh;
  p.W0 = W[0]; p.W1 = W[1]; p.W2 = W[2]; p.W3 = W[3];
  p.bias = bias;
  p.hbf = hbf; p.hf = hf; p.c = c;
  p.G = G;
  p.sync = sync;
  p.out = (float*)d_out;

  mtrnn_persistent<<<NWG, NTHR, 0, stream>>>(p);
}
